// MultiHeadSelfAttention_89945205113263
// MI455X (gfx1250) — matrix operand order repaired and checked by reading, not compiled, not run on hardware
//
#include <hip/hip_runtime.h>

// ---------------------------------------------------------------------------
// MHA forward for MI455X (gfx1250, wave32, WMMA bf16 16x16x32)
// B=2, T=2048, D=1024, H=16, DK=64
// ---------------------------------------------------------------------------

#define B_ 2
#define T_ 2048
#define D_ 1024
#define H_ 16
#define DK_ 64

typedef __bf16 bf16;
typedef bf16  bf16x8_t  __attribute__((ext_vector_type(8)));
typedef bf16  bf16x16_t __attribute__((ext_vector_type(16)));
typedef float f32x8     __attribute__((ext_vector_type(8)));

// D = A(16x32 bf16) * B(32x16 bf16) + C(16x16 f32)
__device__ __forceinline__ f32x8 wmma_bf16(bf16x16_t a, bf16x16_t b, f32x8 c) {
  return __builtin_amdgcn_wmma_f32_16x16x32_bf16(
      /*neg_a=*/false, a, /*neg_b=*/false, b,
      /*c_mod=*/(short)0, c, /*reuse_a=*/false, /*reuse_b=*/false);
}

// A-fragment (16x32, MxK), row-major source with leading dim `ld` (elements).
// ISA layout: lane<16: row M=lane, K in {0..7} (elems 0..7) and {16..23} (elems 8..15);
//             lane>=16: same row, K {8..15} and {24..31}.
template <typename PT>
__device__ __forceinline__ bf16x16_t load_frag_a(PT base, int ld, int lane) {
  const int m  = lane & 15;
  const int kh = lane >> 4;
  PT p = base + (size_t)m * ld + kh * 8;
  union { bf16x16_t v; bf16x8_t h[2]; } u;
  u.h[0] = *(const bf16x8_t*)(p);
  u.h[1] = *(const bf16x8_t*)(p + 16);
  return u.v;
}

// B-fragment (32x16, KxN) where memory holds B^T row-major: row n = contiguous K.
// ISA layout: lane<16: col n=lane, K {0..7} (elems 0..7) and {16..23} (elems 8..15); lane>=16: K {8..15} and {24..31}.
template <typename PT>
__device__ __forceinline__ bf16x16_t load_frag_b(PT base, int ld, int lane) {
  const int n  = lane & 15;
  const int kh = lane >> 4;
  PT p = base + (size_t)n * ld + kh * 8;
  union { bf16x16_t v; bf16x8_t h[2]; } u;
  u.h[0] = *(const bf16x8_t*)(p);
  u.h[1] = *(const bf16x8_t*)(p + 16);
  return u.v;
}

// ---------------------------------------------------------------------------
// f32 -> bf16 conversion
// ---------------------------------------------------------------------------
__global__ void cvt_bf16_kernel(const float* __restrict__ in, bf16* __restrict__ out, int n) {
  int i = blockIdx.x * blockDim.x + threadIdx.x;
  if (i < n) out[i] = (bf16)in[i];
}

// ---------------------------------------------------------------------------
// QKV projection: qkv[m, o] = sum_k x[m,k] * Wqkv[o,k]
// M = B*T = 4096, N = 3*D = 3072, K = 1024.
// Block: 256 thr = 8 waves (4 m-waves x 2 n-waves); block tile 128x128;
// wave tile 32x64 (2x4 WMMA frags). Scatters bf16 Q/K/V into (B,H,T,DK).
// ---------------------------------------------------------------------------
__global__ __launch_bounds__(256)
void qkv_gemm_kernel(const bf16* __restrict__ A, const bf16* __restrict__ W,
                     bf16* __restrict__ Qo, bf16* __restrict__ Ko, bf16* __restrict__ Vo) {
  const int lane = threadIdx.x & 31;
  const int wave = threadIdx.x >> 5;
  const int wm = wave & 3, wn = wave >> 2;
  const int mblk = blockIdx.x * 128 + wm * 32;
  const int nblk = blockIdx.y * 128 + wn * 64;

  f32x8 acc[2][4] = {};
  for (int k0 = 0; k0 < D_; k0 += 32) {
    bf16x16_t a0 = load_frag_a(A + (size_t)mblk * D_ + k0, D_, lane);
    bf16x16_t a1 = load_frag_a(A + (size_t)(mblk + 16) * D_ + k0, D_, lane);
#pragma unroll
    for (int nf = 0; nf < 4; ++nf) {
      bf16x16_t b = load_frag_b(W + (size_t)(nblk + nf * 16) * D_ + k0, D_, lane);
      acc[0][nf] = wmma_bf16(a0, b, acc[0][nf]);
      acc[1][nf] = wmma_bf16(a1, b, acc[1][nf]);
    }
  }

  const int rbase = (lane >> 4) * 8;
  const int nlane = lane & 15;
#pragma unroll
  for (int mf = 0; mf < 2; ++mf) {
#pragma unroll
    for (int nf = 0; nf < 4; ++nf) {
#pragma unroll
      for (int r = 0; r < 8; ++r) {
        int m = mblk + mf * 16 + rbase + r;
        int o = nblk + nf * 16 + nlane;
        int b = m >> 11, t = m & (T_ - 1);
        int which = o >> 10, rem = o & (D_ - 1);
        int h = rem >> 6, d = rem & (DK_ - 1);
        size_t idx = (((size_t)b * H_ + h) * T_ + t) * DK_ + d;
        bf16 v = (bf16)acc[mf][nf][r];
        if (which == 0)      Qo[idx] = v;
        else if (which == 1) Ko[idx] = v;
        else                 Vo[idx] = v;
      }
    }
  }
}

// ---------------------------------------------------------------------------
// RoPE in place on Q and K, layout (B,H,T,DK), interleaved pairs.
// One thread per (b,h,t,pair): i = ((b*H+h) << 16) | (t << 5) | pair
// ---------------------------------------------------------------------------
__global__ void rope_kernel(bf16* __restrict__ Q, bf16* __restrict__ K) {
  unsigned i = blockIdx.x * blockDim.x + threadIdx.x;   // B*H*T*32 total
  int p = i & 31;
  int t = (i >> 5) & (T_ - 1);
  size_t base = (size_t)(i >> 16) * (T_ * DK_) + (size_t)t * DK_ + p * 2;
  float inv = __powf(10000.0f, -(float)(2 * p) / (float)DK_);
  float ang = (float)t * inv;
  float c = __cosf(ang), s = __sinf(ang);
  float q0 = (float)Q[base], q1 = (float)Q[base + 1];
  Q[base]     = (bf16)(q0 * c - q1 * s);
  Q[base + 1] = (bf16)(q0 * s + q1 * c);
  float k0 = (float)K[base], k1 = (float)K[base + 1];
  K[base]     = (bf16)(k0 * c - k1 * s);
  K[base + 1] = (bf16)(k0 * s + k1 * c);
}

// ---------------------------------------------------------------------------
// Causal flash attention. Block = 128 thr (4 waves), each wave owns 16 q rows,
// block owns 64 q rows of one (b,h). Keys tiled by 32.
// K tile staged via GLOBAL_LOAD_ASYNC_TO_LDS_B128 (ASYNCcnt path), V^T staged
// via VGPR transpose; next tile prefetched with global_prefetch_b8.
// P round-trips through per-wave LDS (C-layout -> A-layout) for the PV WMMA.
// Output bf16 (b, t, h*dk).
// ---------------------------------------------------------------------------
__global__ __launch_bounds__(128)
void fmha_kernel(const bf16* __restrict__ Q, const bf16* __restrict__ K,
                 const bf16* __restrict__ V, bf16* __restrict__ Oo) {
  __shared__ __align__(16) bf16 Klds[32][DK_];     // keys x dk
  __shared__ __align__(16) bf16 Vtlds[DK_][32];    // dk x keys (transposed)
  __shared__ __align__(16) bf16 Plds[4][16][32];   // per-wave P tile

  const int lane = threadIdx.x & 31;
  const int wave = threadIdx.x >> 5;
  const int bh = blockIdx.y;                  // b*H + h
  const size_t hb = (size_t)bh * T_ * DK_;
  const int qb = blockIdx.x * 64;
  const int qw = qb + wave * 16;
  const int rbase = (lane >> 4) * 8;
  const int nlane = lane & 15;
  const float scale = 0.125f;                 // 1/sqrt(64)

  // Q fragments for this wave (dk 0..31 and 32..63), held in registers
  bf16x16_t qf0 = load_frag_a(Q + hb + (size_t)qw * DK_ + 0,  DK_, lane);
  bf16x16_t qf1 = load_frag_a(Q + hb + (size_t)qw * DK_ + 32, DK_, lane);

  f32x8 o_acc[4] = {};
  float mrow[8], lrow[8];
#pragma unroll
  for (int r = 0; r < 8; ++r) { mrow[r] = -1e30f; lrow[r] = 0.0f; }

  // per-thread staging coordinates: 128 thr x 16 elems = one 32x64 tile
  const int soff = threadIdx.x * 16;
  const int skk = soff >> 6, sdd = soff & (DK_ - 1);

  const int kend = qb + 64;                   // causal: keys [0, qb+64)
  for (int kt = 0; kt < kend; kt += 32) {
    __syncthreads();
    {
      // K tile (32x64): async DMA straight into LDS (ASYNCcnt-tracked).
      // LDS address for async ops = low 32 bits of the generic pointer
      // (LDS aperture truncation, ISA 10.2); inst offset applies to both
      // the global and LDS sides.
      const bf16* kp = K + hb + (size_t)(kt + skk) * DK_ + sdd;
      unsigned kl = (unsigned)(size_t)&Klds[skk][sdd];
      asm volatile("global_load_async_to_lds_b128 %0, %1, off"
                   :: "v"(kl), "v"(kp) : "memory");
      asm volatile("global_load_async_to_lds_b128 %0, %1, off offset:16"
                   :: "v"(kl), "v"(kp) : "memory");

      // V tile (32x64) staged transposed through VGPRs
      const bf16* vp = V + hb + (size_t)(kt + skk) * DK_ + sdd;
      bf16x8_t v0 = *(const bf16x8_t*)(vp);
      bf16x8_t v1 = *(const bf16x8_t*)(vp + 8);
#pragma unroll
      for (int j = 0; j < 8; ++j) {
        Vtlds[sdd + j][skk]     = v0[j];
        Vtlds[sdd + 8 + j][skk] = v1[j];
      }

      // prefetch next key tile while this one is consumed
      if (kt + 32 < kend) {
        __builtin_prefetch(kp + 32 * DK_, 0, 3);
        __builtin_prefetch(vp + 32 * DK_, 0, 3);
      }
      asm volatile("s_wait_asynccnt 0x0" ::: "memory");
    }
    __syncthreads();

    // S = Q K^T for two 16-key sub-tiles (contraction over dk = 64)
    f32x8 s0 = {}, s1 = {};
    {
      bf16x16_t kb0  = load_frag_b(&Klds[0][0],   DK_, lane);
      bf16x16_t kb0b = load_frag_b(&Klds[0][32],  DK_, lane);
      s0 = wmma_bf16(qf0, kb0,  s0);
      s0 = wmma_bf16(qf1, kb0b, s0);
      bf16x16_t kb1  = load_frag_b(&Klds[16][0],  DK_, lane);
      bf16x16_t kb1b = load_frag_b(&Klds[16][32], DK_, lane);
      s1 = wmma_bf16(qf0, kb1,  s1);
      s1 = wmma_bf16(qf1, kb1b, s1);
    }

    // causal mask + scale + online softmax (row reductions over 16 lanes)
#pragma unroll
    for (int r = 0; r < 8; ++r) {
      int qg = qw + rbase + r;
      float a = (kt + nlane      <= qg) ? s0[r] * scale : -1e30f;
      float b = (kt + 16 + nlane <= qg) ? s1[r] * scale : -1e30f;
      float mx = fmaxf(a, b);
      mx = fmaxf(mx, __shfl_xor(mx, 1, 32));
      mx = fmaxf(mx, __shfl_xor(mx, 2, 32));
      mx = fmaxf(mx, __shfl_xor(mx, 4, 32));
      mx = fmaxf(mx, __shfl_xor(mx, 8, 32));
      float mnew = fmaxf(mrow[r], mx);
      float al = __expf(mrow[r] - mnew);
      float p0 = __expf(a - mnew);
      float p1 = __expf(b - mnew);
      float rs = p0 + p1;
      rs += __shfl_xor(rs, 1, 32);
      rs += __shfl_xor(rs, 2, 32);
      rs += __shfl_xor(rs, 4, 32);
      rs += __shfl_xor(rs, 8, 32);
      lrow[r] = lrow[r] * al + rs;
      mrow[r] = mnew;
#pragma unroll
      for (int nf = 0; nf < 4; ++nf) o_acc[nf][r] *= al;
      Plds[wave][rbase + r][nlane]      = (bf16)p0;   // C-layout -> LDS
      Plds[wave][rbase + r][16 + nlane] = (bf16)p1;
    }

    // ensure per-wave P stores are visible before A-layout reload
    asm volatile("s_wait_dscnt 0x0" ::: "memory");

    // O += P(16x32) @ V(32x64)
    bf16x16_t pf = load_frag_a(&Plds[wave][0][0], 32, lane);
#pragma unroll
    for (int nf = 0; nf < 4; ++nf) {
      bf16x16_t vf = load_frag_b(&Vtlds[nf * 16][0], 32, lane);
      o_acc[nf] = wmma_bf16(pf, vf, o_acc[nf]);
    }
  }

  // epilogue: O /= l, write bf16 to (b, t, h*dk)
  const int b = bh >> 4, h = bh & 15;
#pragma unroll
  for (int nf = 0; nf < 4; ++nf) {
#pragma unroll
    for (int r = 0; r < 8; ++r) {
      int qg = qw + rbase + r;
      float val = o_acc[nf][r] / lrow[r];
      size_t idx = ((size_t)b * T_ + qg) * D_ + h * DK_ + nf * 16 + nlane;
      Oo[idx] = (bf16)val;
    }
  }
}

// ---------------------------------------------------------------------------
// Output projection: out[m, o] = sum_k attn[m,k] * Wout[o,k]  (f32 output)
// M = 4096, N = 1024, K = 1024. Same tiling as QKV GEMM.
// ---------------------------------------------------------------------------
__global__ __launch_bounds__(256)
void outproj_gemm_kernel(const bf16* __restrict__ A, const bf16* __restrict__ W,
                         float* __restrict__ out) {
  const int lane = threadIdx.x & 31;
  const int wave = threadIdx.x >> 5;
  const int wm = wave & 3, wn = wave >> 2;
  const int mblk = blockIdx.x * 128 + wm * 32;
  const int nblk = blockIdx.y * 128 + wn * 64;

  f32x8 acc[2][4] = {};
  for (int k0 = 0; k0 < D_; k0 += 32) {
    bf16x16_t a0 = load_frag_a(A + (size_t)mblk * D_ + k0, D_, lane);
    bf16x16_t a1 = load_frag_a(A + (size_t)(mblk + 16) * D_ + k0, D_, lane);
#pragma unroll
    for (int nf = 0; nf < 4; ++nf) {
      bf16x16_t b = load_frag_b(W + (size_t)(nblk + nf * 16) * D_ + k0, D_, lane);
      acc[0][nf] = wmma_bf16(a0, b, acc[0][nf]);
      acc[1][nf] = wmma_bf16(a1, b, acc[1][nf]);
    }
  }

  const int rbase = (lane >> 4) * 8;
  const int nlane = lane & 15;
#pragma unroll
  for (int mf = 0; mf < 2; ++mf) {
#pragma unroll
    for (int nf = 0; nf < 4; ++nf) {
#pragma unroll
      for (int r = 0; r < 8; ++r) {
        int m = mblk + mf * 16 + rbase + r;
        int o = nblk + nf * 16 + nlane;
        out[(size_t)m * D_ + o] = acc[mf][nf][r];
      }
    }
  }
}

// ---------------------------------------------------------------------------
// launcher
// ---------------------------------------------------------------------------
extern "C" void kernel_launch(void* const* d_in, const int* in_sizes, int n_in,
                              void* d_out, int out_size, void* d_ws, size_t ws_size,
                              hipStream_t stream) {
  (void)in_sizes; (void)n_in; (void)out_size; (void)ws_size;
  const float* x    = (const float*)d_in[0];   // (B,T,D)
  const float* Wqkv = (const float*)d_in[1];   // (3D,D)
  const float* Wout = (const float*)d_in[2];   // (D,D)
  float* out = (float*)d_out;                  // (B,T,D)

  const size_t nX    = (size_t)B_ * T_ * D_;   // 4,194,304
  const size_t nWqkv = (size_t)3 * D_ * D_;    // 3,145,728
  const size_t nWout = (size_t)D_ * D_;        // 1,048,576
  const size_t nHead = (size_t)B_ * H_ * T_ * DK_; // 4,194,304

  bf16* xb    = (bf16*)d_ws;
  bf16* wqkvb = xb + nX;
  bf16* woutb = wqkvb + nWqkv;
  bf16* Qb    = woutb + nWout;
  bf16* Kb    = Qb + nHead;
  bf16* Vb    = Kb + nHead;
  bf16* attn  = Vb + nHead;                    // (B*T, D) bf16

  cvt_bf16_kernel<<<(int)(nX    / 256), 256, 0, stream>>>(x,    xb,    (int)nX);
  cvt_bf16_kernel<<<(int)(nWqkv / 256), 256, 0, stream>>>(Wqkv, wqkvb, (int)nWqkv);
  cvt_bf16_kernel<<<(int)(nWout / 256), 256, 0, stream>>>(Wout, woutb, (int)nWout);

  qkv_gemm_kernel<<<dim3(32, 24), 256, 0, stream>>>(xb, wqkvb, Qb, Kb, Vb);

  rope_kernel<<<(int)((size_t)B_ * H_ * T_ * (DK_ / 2) / 256), 256, 0, stream>>>(Qb, Kb);

  fmha_kernel<<<dim3(T_ / 64, B_ * H_), 128, 0, stream>>>(Qb, Kb, Vb, attn);

  outproj_gemm_kernel<<<dim3(32, 8), 256, 0, stream>>>(attn, woutb, out);
}
